// DirGINELayer_90589450207805
// MI455X (gfx1250) — matrix operand order repaired and checked by reading, not compiled, not run on hardware
//
#include <hip/hip_runtime.h>

#define NNODES 50000
#define EDIR   800000
#define ETOT   1600000
#define IND    128
#define OUTD   128
#define EAD    16
#define REQD   16
#define CATD   160

typedef __bf16 bf16;
typedef bf16  v16bf __attribute__((ext_vector_type(16)));
typedef bf16  v8bf  __attribute__((ext_vector_type(8)));
typedef float v8f   __attribute__((ext_vector_type(8)));

// Build a 16-element bf16 fragment from two 8-element (16B) LDS loads.
__device__ __forceinline__ v16bf make_frag(const bf16* p0, const bf16* p1) {
  v8bf lo = *(const v8bf*)p0;
  v8bf hi = *(const v8bf*)p1;
  v16bf r;
#pragma unroll
  for (int i = 0; i < 8; ++i) { r[i] = lo[i]; r[i + 8] = hi[i]; }
  return r;
}

// ---------------- prep kernels ----------------
__global__ void cvt_x_kernel(const float* __restrict__ in, bf16* __restrict__ out, int n) {
  for (int i = blockIdx.x * blockDim.x + threadIdx.x; i < n; i += gridDim.x * blockDim.x)
    out[i] = (bf16)in[i];
}

// WT[n*K + k] = W[k*Nn + n]   (W is K x Nn row-major)
__global__ void transpose_w_kernel(const float* __restrict__ W, bf16* __restrict__ WT,
                                   int K, int Nn) {
  int i = blockIdx.x * blockDim.x + threadIdx.x;
  if (i < K * Nn) {
    int n = i / K, k = i % K;
    WT[n * K + k] = (bf16)W[k * Nn + n];
  }
}

__global__ void cvt_req_kernel(const float* __restrict__ r, bf16* __restrict__ o) {
  if (threadIdx.x < REQD) o[threadIdx.x] = (bf16)r[threadIdx.x];
}

// ---------------- edge message kernel ----------------
// Per WG: 8 waves x 32 edges = 256 edges. Each wave owns two 16-row M tiles so
// every B fragment (LDS) feeds 2 WMMAs -> ~1.25 ds_load_b128 per WMMA.
// W1T:[128][160] bf16, W2T:[128][128] bf16.
__global__ __launch_bounds__(256)
void edge_msg_kernel(const bf16* __restrict__ xb,
                     const int* __restrict__ srcIdx,
                     const int* __restrict__ dstIdx,
                     const float* __restrict__ eattr,
                     const bf16* __restrict__ req16,
                     const bf16* __restrict__ W1T,
                     const float* __restrict__ b1,
                     const bf16* __restrict__ W2T,
                     const float* __restrict__ b2,
                     float* __restrict__ acc) {
  extern __shared__ char smem[];
  bf16* sW1 = (bf16*)smem;              // 128*160
  bf16* sW2 = sW1 + 128 * CATD;         // 128*128
  bf16* sA  = sW2 + 128 * OUTD;         // 8 waves * 32 * 160 (reused as h1 [32][128])
  int*  sDst = (int*)(sA + 8 * 32 * CATD);  // 256

  const int tid  = threadIdx.x;
  const int wave = tid >> 5;
  const int lane = tid & 31;

  // stage weights into LDS (16B chunks)
  {
    const uint4* g1 = (const uint4*)W1T;
    uint4* s1 = (uint4*)sW1;
    for (int i = tid; i < (128 * CATD) / 8; i += 256) s1[i] = g1[i];
    const uint4* g2 = (const uint4*)W2T;
    uint4* s2 = (uint4*)sW2;
    for (int i = tid; i < (128 * OUTD) / 8; i += 256) s2[i] = g2[i];
  }

  // build per-wave A tile: 32 rows x [x_j(128) | edge_attr(16) | req(16)]
  // one lane per edge row
  const int ebase = blockIdx.x * 256 + wave * 32;
  bf16* aw = sA + wave * 32 * CATD;
  {
    const int e = ebase + lane;
    const int s = srcIdx[e];
    const uint4* xrow = (const uint4*)(xb + (size_t)s * IND);
    uint4* arow = (uint4*)(aw + lane * CATD);
#pragma unroll
    for (int i = 0; i < 16; ++i) arow[i] = xrow[i];
    const float* ea = eattr + (size_t)e * EAD;
    bf16* p = aw + lane * CATD + IND;
#pragma unroll
    for (int i = 0; i < EAD; ++i) p[i] = (bf16)ea[i];
    uint4* q = (uint4*)(aw + lane * CATD + IND + EAD);
    const uint4* rq = (const uint4*)req16;
    q[0] = rq[0];
    q[1] = rq[1];
    sDst[wave * 32 + lane] = dstIdx[e];
  }
  __syncthreads();

  const int colq  = lane & 15;         // N within 16-tile / M row for A
  const int rbase = (lane >> 4) * 8;   // C/D row base
  const int kbA   = (lane >> 4) * 8;   // A-frag K sub-offset
  const int kbB   = (lane >> 4) * 8;   // B-frag K sub-offset

  v8f c0[8], c1[8];
#pragma unroll
  for (int nt = 0; nt < 8; ++nt)
#pragma unroll
    for (int i = 0; i < 8; ++i) { c0[nt][i] = 0.0f; c1[nt][i] = 0.0f; }

  // GEMM1: (32x160) @ (160x128), K-steps of 32, two M tiles share each B frag
#pragma unroll
  for (int ks = 0; ks < 5; ++ks) {
    const int k0 = ks * 32;
    const bf16* ap0 = aw + colq * CATD + k0 + kbA;
    const bf16* ap1 = aw + (16 + colq) * CATD + k0 + kbA;
    v16bf a0 = make_frag(ap0, ap0 + 16);
    v16bf a1 = make_frag(ap1, ap1 + 16);
#pragma unroll
    for (int nt = 0; nt < 8; ++nt) {
      const bf16* bp = sW1 + (nt * 16 + colq) * CATD + k0 + kbB;
      v16bf b = make_frag(bp, bp + 16);
      c0[nt] = __builtin_amdgcn_wmma_f32_16x16x32_bf16(false, a0, false, b,
                                                       (short)0, c0[nt], false, false);
      c1[nt] = __builtin_amdgcn_wmma_f32_16x16x32_bf16(false, a1, false, b,
                                                       (short)0, c1[nt], false, false);
    }
  }

  // bias + relu -> bf16 h1 tile (reuse A-tile LDS space, [32][128])
  bf16* hw = aw;
#pragma unroll
  for (int nt = 0; nt < 8; ++nt) {
    const int col = nt * 16 + colq;
    const float bv = b1[col];
#pragma unroll
    for (int v = 0; v < 8; ++v) {
      float t0 = c0[nt][v] + bv;
      t0 = t0 > 0.0f ? t0 : 0.0f;
      hw[(rbase + v) * OUTD + col] = (bf16)t0;
      float t1 = c1[nt][v] + bv;
      t1 = t1 > 0.0f ? t1 : 0.0f;
      hw[(16 + rbase + v) * OUTD + col] = (bf16)t1;
    }
#pragma unroll
    for (int i = 0; i < 8; ++i) { c0[nt][i] = 0.0f; c1[nt][i] = 0.0f; }
  }

  // GEMM2: (32x128) @ (128x128)
#pragma unroll
  for (int ks = 0; ks < 4; ++ks) {
    const int k0 = ks * 32;
    const bf16* ap0 = hw + colq * OUTD + k0 + kbA;
    const bf16* ap1 = hw + (16 + colq) * OUTD + k0 + kbA;
    v16bf a0 = make_frag(ap0, ap0 + 16);
    v16bf a1 = make_frag(ap1, ap1 + 16);
#pragma unroll
    for (int nt = 0; nt < 8; ++nt) {
      const bf16* bp = sW2 + (nt * 16 + colq) * OUTD + k0 + kbB;
      v16bf b = make_frag(bp, bp + 16);
      c0[nt] = __builtin_amdgcn_wmma_f32_16x16x32_bf16(false, a0, false, b,
                                                       (short)0, c0[nt], false, false);
      c1[nt] = __builtin_amdgcn_wmma_f32_16x16x32_bf16(false, a1, false, b,
                                                       (short)0, c1[nt], false, false);
    }
  }

  // epilogue: bias + scatter-add into node accumulator (L2-resident atomics)
#pragma unroll
  for (int nt = 0; nt < 8; ++nt) {
    const int col = nt * 16 + colq;
    const float bv = b2[col];
#pragma unroll
    for (int v = 0; v < 8; ++v) {
      const int n0 = sDst[wave * 32 + rbase + v];
      atomicAdd(acc + (size_t)n0 * OUTD + col, c0[nt][v] + bv);
      const int n1 = sDst[wave * 32 + 16 + rbase + v];
      atomicAdd(acc + (size_t)n1 * OUTD + col, c1[nt][v] + bv);
    }
  }
}

// ---------------- node update kernel ----------------
// out = relu(((1+eps)*x + acc) @ Wu + bu), computed in place over d_out.
__global__ __launch_bounds__(256)
void node_update_kernel(const float* __restrict__ x,
                        const float* acc_in,        // aliases out
                        const bf16* __restrict__ WuT,
                        const float* __restrict__ bu,
                        const float* __restrict__ eps,
                        float* out) {
  extern __shared__ char smem[];
  bf16* sWu = (bf16*)smem;          // 128*128
  bf16* sA  = sWu + 128 * OUTD;     // 8 waves * 16 * 128

  const int tid = threadIdx.x, wave = tid >> 5, lane = tid & 31;
  {
    const uint4* g = (const uint4*)WuT;
    uint4* s = (uint4*)sWu;
    for (int i = tid; i < (128 * OUTD) / 8; i += 256) s[i] = g[i];
  }
  __syncthreads();

  const int t = blockIdx.x * 8 + wave;   // 16-row tile index
  if (t >= NNODES / 16) return;

  const float epsv = 1.0f + eps[0];
  bf16* aw = sA + wave * 16 * OUTD;
  {
    const int r = lane >> 1, hh = lane & 1;
    const size_t g = (size_t)(t * 16 + r) * OUTD + hh * 64;
    const float4* xp = (const float4*)(x + g);
    const float4* hp = (const float4*)(acc_in + g);
    bf16* ap = aw + r * OUTD + hh * 64;
#pragma unroll
    for (int i = 0; i < 16; ++i) {
      float4 xv = xp[i], hv = hp[i];
      ap[4 * i + 0] = (bf16)(epsv * xv.x + hv.x);
      ap[4 * i + 1] = (bf16)(epsv * xv.y + hv.y);
      ap[4 * i + 2] = (bf16)(epsv * xv.z + hv.z);
      ap[4 * i + 3] = (bf16)(epsv * xv.w + hv.w);
    }
  }

  const int colq  = lane & 15;
  const int rbase = (lane >> 4) * 8;
  const int kbA   = (lane >> 4) * 8;
  const int kbB   = (lane >> 4) * 8;

  v8f c[8];
#pragma unroll
  for (int nt = 0; nt < 8; ++nt)
#pragma unroll
    for (int i = 0; i < 8; ++i) c[nt][i] = 0.0f;

#pragma unroll
  for (int ks = 0; ks < 4; ++ks) {
    const int k0 = ks * 32;
    const bf16* ap = aw + colq * OUTD + k0 + kbA;
    v16bf a = make_frag(ap, ap + 16);
#pragma unroll
    for (int nt = 0; nt < 8; ++nt) {
      const bf16* bp = sWu + (nt * 16 + colq) * OUTD + k0 + kbB;
      v16bf b = make_frag(bp, bp + 16);
      c[nt] = __builtin_amdgcn_wmma_f32_16x16x32_bf16(false, a, false, b,
                                                      (short)0, c[nt], false, false);
    }
  }

#pragma unroll
  for (int nt = 0; nt < 8; ++nt) {
    const int col = nt * 16 + colq;
    const float bv = bu[col];
#pragma unroll
    for (int v = 0; v < 8; ++v) {
      float tv = c[nt][v] + bv;
      tv = tv > 0.0f ? tv : 0.0f;
      out[(size_t)(t * 16 + rbase + v) * OUTD + col] = tv;
    }
  }
}

extern "C" void kernel_launch(void* const* d_in, const int* in_sizes, int n_in,
                              void* d_out, int out_size, void* d_ws, size_t ws_size,
                              hipStream_t stream) {
  const float* x   = (const float*)d_in[0];
  const int*   ei  = (const int*)d_in[1];
  const float* ea  = (const float*)d_in[2];
  const float* req = (const float*)d_in[3];
  const float* W1f = (const float*)d_in[4];
  const float* b1f = (const float*)d_in[5];
  const float* W2f = (const float*)d_in[6];
  const float* b2f = (const float*)d_in[7];
  const float* W1b = (const float*)d_in[8];
  const float* b1b = (const float*)d_in[9];
  const float* W2b = (const float*)d_in[10];
  const float* b2b = (const float*)d_in[11];
  const float* Wu  = (const float*)d_in[12];
  const float* bu  = (const float*)d_in[13];
  const float* eps = (const float*)d_in[14];

  char* ws = (char*)d_ws;
  size_t off = 0;
  auto walloc = [&](size_t bytes) {
    void* p = ws + off;
    off = (off + bytes + 255) & ~(size_t)255;
    return p;
  };
  bf16* xb   = (bf16*)walloc((size_t)NNODES * IND * 2);
  bf16* W1fT = (bf16*)walloc((size_t)CATD * OUTD * 2);
  bf16* W1bT = (bf16*)walloc((size_t)CATD * OUTD * 2);
  bf16* W2fT = (bf16*)walloc((size_t)OUTD * OUTD * 2);
  bf16* W2bT = (bf16*)walloc((size_t)OUTD * OUTD * 2);
  bf16* WuT  = (bf16*)walloc((size_t)OUTD * OUTD * 2);
  bf16* reqb = (bf16*)walloc(256);

  // zero node accumulator (d_out doubles as the segment-sum accumulator)
  hipMemsetAsync(d_out, 0, (size_t)NNODES * OUTD * sizeof(float), stream);

  cvt_x_kernel<<<2048, 256, 0, stream>>>(x, xb, NNODES * IND);
  transpose_w_kernel<<<(CATD * OUTD + 255) / 256, 256, 0, stream>>>(W1f, W1fT, CATD, OUTD);
  transpose_w_kernel<<<(CATD * OUTD + 255) / 256, 256, 0, stream>>>(W1b, W1bT, CATD, OUTD);
  transpose_w_kernel<<<(OUTD * OUTD + 255) / 256, 256, 0, stream>>>(W2f, W2fT, OUTD, OUTD);
  transpose_w_kernel<<<(OUTD * OUTD + 255) / 256, 256, 0, stream>>>(W2b, W2bT, OUTD, OUTD);
  transpose_w_kernel<<<(OUTD * OUTD + 255) / 256, 256, 0, stream>>>(Wu, WuT, OUTD, OUTD);
  cvt_req_kernel<<<1, 32, 0, stream>>>(req, reqb);

  const size_t smem_edge =
      (size_t)(128 * CATD + 128 * OUTD + 8 * 32 * CATD) * 2 + 256 * 4;  // 156672 B
  // forward direction: src = ei[0,:E], dst = ei[1,:E]
  edge_msg_kernel<<<EDIR / 256, 256, smem_edge, stream>>>(
      xb, ei, ei + ETOT, ea, reqb, W1fT, b1f, W2fT, b2f, (float*)d_out);
  // backward direction: src = ei[0,E:], dst = ei[1,E:]
  edge_msg_kernel<<<EDIR / 256, 256, smem_edge, stream>>>(
      xb, ei + EDIR, ei + ETOT + EDIR, ea + (size_t)EDIR * EAD, reqb,
      W1bT, b1b, W2bT, b2b, (float*)d_out);

  const size_t smem_node = (size_t)(128 * OUTD + 8 * 16 * OUTD) * 2;  // 65536 B
  node_update_kernel<<<(NNODES / 16 + 7) / 8, 256, smem_node, stream>>>(
      x, (const float*)d_out, WuT, bu, eps, (float*)d_out);
}
